// Graphormer_45844480918020
// MI455X (gfx1250) — matrix operand order repaired and checked by reading, not compiled, not run on hardware
//
#include <hip/hip_runtime.h>
#include <hip/hip_bf16.h>
#include <math.h>

typedef __attribute__((ext_vector_type(16))) _Float16 v16h;
typedef __attribute__((ext_vector_type(8)))  float    v8f;

#define GB   128      // batch B
#define GS   128      // nodes per graph S (incl. virtual node)
#define GH   512      // hidden H
#define GNH  8        // heads
#define GDH  64       // head dim
#define GNL  4        // layers
#define GFFN 2048
#define GM   (GB*GS)  // 16384 rows
#define GFN  128      // F_NODE
#define GE   131072   // edges
#define GPP  (GB*GS*GS)
#define GLP  5        // path len

// ---------------- fragment helpers (CDNA5 16-bit WMMA layouts) ----------------
union FragH { v16h v; int4 q[2]; unsigned int u[8]; };
union FragF { v8f  v; float        f[8]; };
union PairH { unsigned int u; _Float16 h[2]; };

__device__ __forceinline__ void fzero(FragF& a) {
#pragma unroll
  for (int i = 0; i < 8; ++i) a.f[i] = 0.f;
}

// A fragment: 16 rows x 32 K, f16, from row-major tile [row][k] with given stride (halfs).
// lane<16: rows 0..15, K groups {0..7,16..23}; lane>=16: same rows, K {8..15,24..31}.
// Per lane this is two contiguous 16-byte chunks -> 2x ds_load_b128.
__device__ __forceinline__ v16h frag_a(const _Float16* t, int row0, int stride, int k0) {
  int lane = threadIdx.x & 31;
  const _Float16* p = t + (size_t)(row0 + (lane & 15)) * stride + k0 + ((lane >> 4) << 3);
  FragH f;
  f.q[0] = *(const int4*)(p);        // halfs k..k+7
  f.q[1] = *(const int4*)(p + 16);   // halfs k+16..k+23
  return f.v;
}

// B fragment: 32 K x 16 cols, f16, from K-transposed tile [col][k] with given stride (halfs).
// lane&15 = col; lane<16: K {0..7,16..23}; lane>=16: K {8..15,24..31} -> 2x ds_load_b128.
__device__ __forceinline__ v16h frag_b(const _Float16* t, int col0, int stride, int k0) {
  int lane = threadIdx.x & 31;
  const _Float16* p = t + (size_t)(col0 + (lane & 15)) * stride + k0 + ((lane >> 4) << 3);
  FragH f;
  f.q[0] = *(const int4*)(p);        // halfs k..k+7
  f.q[1] = *(const int4*)(p + 16);   // halfs k+16..k+23
  return f.v;
}

// Branch-free tanh-approx GELU: tanh(u) = 1 - 2/(exp(2u)+1)  (exact limits at +/-inf)
__device__ __forceinline__ float gelu_f(float x) {
  float u = 0.7978845608028654f * (x + 0.044715f * x * x * x);
  float e = __expf(2.f * u);
  float t = 1.f - 2.f / (e + 1.f);
  return 0.5f * x * (1.f + t);
}

// ---------------- small utility kernels ----------------
__global__ void k_f32_to_f16(const float* __restrict__ s, _Float16* __restrict__ d, int n) {
  int i = blockIdx.x * blockDim.x + threadIdx.x;
  if (i < n) d[i] = (_Float16)s[i];
}

__global__ void k_zero_i32(int* p, int n) {
  int i = blockIdx.x * blockDim.x + threadIdx.x;
  if (i < n) p[i] = 0;
}

__global__ void k_degrees(const int* __restrict__ ei, int* __restrict__ outdeg, int* __restrict__ indeg) {
  int e = blockIdx.x * blockDim.x + threadIdx.x;
  if (e < GE) {
    atomicAdd(&outdeg[ei[e]], 1);       // row 0 = src
    atomicAdd(&indeg[ei[GE + e]], 1);   // row 1 = dst
  }
}

// Build f16 input matrix (16384 x 128) including virtual-node rows of -1.0
__global__ void k_build_x16(const float* __restrict__ x, _Float16* __restrict__ x16) {
  int i = blockIdx.x * blockDim.x + threadIdx.x;
  if (i >= GM * GFN) return;
  int r = i >> 7, kk = i & 127;
  int s = r & (GS - 1), bg = r >> 7;
  float v = (s == 0) ? -1.0f : x[(size_t)(bg * (GS - 1) + s - 1) * GFN + kk];
  x16[i] = (_Float16)v;
}

__global__ void k_degadd(float* __restrict__ h, const int* __restrict__ indeg,
                         const int* __restrict__ outdeg,
                         const float* __restrict__ z_in, const float* __restrict__ z_out) {
  int i = blockIdx.x * blockDim.x + threadIdx.x;
  if (i >= GM * GH) return;
  int node = i >> 9, j = i & (GH - 1);
  int di = indeg[node];  if (di > 63) di = 63;
  int dd = outdeg[node]; if (dd > 63) dd = 63;
  h[i] += z_in[(size_t)di * GH + j] + z_out[(size_t)dd * GH + j];
}

// Fold W_edge (16x64) @ edge_vector^T (64x5) into Wc (16x5), bc (5)
__global__ void k_wc(const float* __restrict__ We, const float* __restrict__ be,
                     const float* __restrict__ ev, float* __restrict__ wc) {
  int t = threadIdx.x;
  if (t < 80) {
    int kk = t / 5, p = t % 5;
    float s = 0.f;
    for (int j = 0; j < 64; ++j) s += We[kk * 64 + j] * ev[p * 64 + j];
    wc[t] = s;
  } else if (t < 85) {
    int p = t - 80;
    float s = 0.f;
    for (int j = 0; j < 64; ++j) s += be[j] * ev[p * 64 + j];
    wc[80 + p] = s;
  }
}

__global__ void k_stepscores(const float* __restrict__ ea, const float* __restrict__ wc,
                             float* __restrict__ ss) {
  int e = blockIdx.x * blockDim.x + threadIdx.x;
  if (e >= GE) return;
  float a[16];
#pragma unroll
  for (int k = 0; k < 16; ++k) a[k] = ea[(size_t)e * 16 + k];
#pragma unroll
  for (int p = 0; p < 5; ++p) {
    float s = wc[80 + p];
#pragma unroll
    for (int k = 0; k < 16; ++k) s += a[k] * wc[k * 5 + p];
    ss[(size_t)e * 5 + p] = s;
  }
}

__global__ void k_bias(const int* __restrict__ npath, const int* __restrict__ epath,
                       const float* __restrict__ ss, const float* __restrict__ bsp,
                       float* __restrict__ bias) {
  int i = blockIdx.x * blockDim.x + threadIdx.x;
  if (i >= GPP) return;
  int nl = 0;
#pragma unroll
  for (int j = 0; j < GLP; ++j) nl += (npath[(size_t)i * GLP + j] != -1);
  int ci = nl - 1; if (ci < 0) ci = 0; if (ci > 4) ci = 4;
  float sp = (nl > 0) ? bsp[ci] : 0.f;
  int el = 0; float s = 0.f;
#pragma unroll
  for (int j = 0; j < GLP; ++j) {
    int ep = epath[(size_t)i * GLP + j];
    if (ep != -1) { ++el; s += ss[(size_t)ep * GLP + j]; }
  }
  float ee = (el > 0) ? s / (float)el : 0.f;
  bias[i] = sp + ee;
}

// LayerNorm over 512-wide rows, fp32 in -> f16 out
__global__ void k_ln(const float* __restrict__ x, const float* __restrict__ g,
                     const float* __restrict__ b, _Float16* __restrict__ y) {
  __shared__ float red[256];
  int row = blockIdx.x, tid = threadIdx.x;
  const float* xr = x + (size_t)row * GH;
  float a = xr[tid], c = xr[tid + 256];
  red[tid] = a + c;
  __syncthreads();
  for (int s = 128; s > 0; s >>= 1) { if (tid < s) red[tid] += red[tid + s]; __syncthreads(); }
  float mean = red[0] * (1.f / GH);
  __syncthreads();
  float da = a - mean, dc = c - mean;
  red[tid] = da * da + dc * dc;
  __syncthreads();
  for (int s = 128; s > 0; s >>= 1) { if (tid < s) red[tid] += red[tid + s]; __syncthreads(); }
  float rstd = rsqrtf(red[0] * (1.f / GH) + 1e-5f);
  _Float16* yr = y + (size_t)row * GH;
  yr[tid]       = (_Float16)(da * rstd * g[tid]       + b[tid]);
  yr[tid + 256] = (_Float16)(dc * rstd * g[tid + 256] + b[tid + 256]);
}

// ---------------- generic f16 WMMA GEMM: C = A(MxK) @ B(KxN) + bias [+resid] [gelu] ----------------
// 256 threads = 8 waves; block tile 128x128; wave tile 32x64 (2x4 WMMA 16x16); K step 32.
// Compile-time flags keep the epilogue branch-free.
template <bool RESID, int ACT, bool OUTH>
__global__ __launch_bounds__(256) void k_gemm(const _Float16* __restrict__ A,
                                              const _Float16* __restrict__ B,
                                              const float* __restrict__ bias,
                                              const float* __restrict__ resid,
                                              float* __restrict__ outF,
                                              _Float16* __restrict__ outH,
                                              int M, int N, int K) {
  __shared__ _Float16 As[128 * 32];   // [row][k]
  __shared__ _Float16 Bs[128 * 32];   // transposed: [n][k]
  int tid = threadIdx.x;
  int lane = tid & 31, w = tid >> 5;
  int m0 = blockIdx.x * 128, n0 = blockIdx.y * 128;
  int wm = (w >> 1) * 32, wn = (w & 1) * 64;

  FragF acc[2][4];
#pragma unroll
  for (int tm = 0; tm < 2; ++tm)
#pragma unroll
    for (int tn = 0; tn < 4; ++tn) fzero(acc[tm][tn]);

  for (int k0 = 0; k0 < K; k0 += 32) {
    __syncthreads();
    // stage A: 128x32 as 2048 dword pairs (coalesced along K)
#pragma unroll 2
    for (int i = tid; i < 2048; i += 256) {
      int row = i >> 4, kp = i & 15;
      *(unsigned int*)&As[row * 32 + kp * 2] =
          *(const unsigned int*)&A[(size_t)(m0 + row) * K + k0 + kp * 2];
    }
    // stage B transposed: read dword pairs coalesced along N, scatter to [n][k]
#pragma unroll 2
    for (int i = tid; i < 2048; i += 256) {
      int kk = i >> 6, np = i & 63;
      PairH pr;
      pr.u = *(const unsigned int*)&B[(size_t)(k0 + kk) * N + n0 + np * 2];
      Bs[(np * 2) * 32 + kk]     = pr.h[0];
      Bs[(np * 2 + 1) * 32 + kk] = pr.h[1];
    }
    __syncthreads();
    v16h fa0 = frag_a(As, wm, 32, 0);
    v16h fa1 = frag_a(As, wm + 16, 32, 0);
    v16h fb0 = frag_b(Bs, wn, 32, 0);
    v16h fb1 = frag_b(Bs, wn + 16, 32, 0);
    v16h fb2 = frag_b(Bs, wn + 32, 32, 0);
    v16h fb3 = frag_b(Bs, wn + 48, 32, 0);
    acc[0][0].v = __builtin_amdgcn_wmma_f32_16x16x32_f16(false, fa0, false, fb0, (short)0, acc[0][0].v, false, false);
    acc[0][1].v = __builtin_amdgcn_wmma_f32_16x16x32_f16(false, fa0, false, fb1, (short)0, acc[0][1].v, false, false);
    acc[0][2].v = __builtin_amdgcn_wmma_f32_16x16x32_f16(false, fa0, false, fb2, (short)0, acc[0][2].v, false, false);
    acc[0][3].v = __builtin_amdgcn_wmma_f32_16x16x32_f16(false, fa0, false, fb3, (short)0, acc[0][3].v, false, false);
    acc[1][0].v = __builtin_amdgcn_wmma_f32_16x16x32_f16(false, fa1, false, fb0, (short)0, acc[1][0].v, false, false);
    acc[1][1].v = __builtin_amdgcn_wmma_f32_16x16x32_f16(false, fa1, false, fb1, (short)0, acc[1][1].v, false, false);
    acc[1][2].v = __builtin_amdgcn_wmma_f32_16x16x32_f16(false, fa1, false, fb2, (short)0, acc[1][2].v, false, false);
    acc[1][3].v = __builtin_amdgcn_wmma_f32_16x16x32_f16(false, fa1, false, fb3, (short)0, acc[1][3].v, false, false);
  }

#pragma unroll
  for (int tm = 0; tm < 2; ++tm) {
#pragma unroll
    for (int tn = 0; tn < 4; ++tn) {
      int n = n0 + wn + tn * 16 + (lane & 15);
      int mb = m0 + wm + tm * 16 + ((lane >> 4) << 3);
      float bv = bias[n];
      float rv[8];
      if constexpr (RESID) {
#pragma unroll
        for (int i = 0; i < 8; ++i) rv[i] = resid[(size_t)(mb + i) * N + n];
      }
#pragma unroll
      for (int i = 0; i < 8; ++i) {
        float val = acc[tm][tn].f[i] + bv;
        if constexpr (RESID) val += rv[i];
        if constexpr (ACT == 1) val = gelu_f(val);
        if constexpr (OUTH) outH[(size_t)(mb + i) * N + n] = (_Float16)val;
        else                outF[(size_t)(mb + i) * N + n] = val;
      }
    }
  }
}

// ---------------- fused attention per (batch, head) ----------------
// grid (B, NH), 256 threads. Dynamic LDS: Q(16K) K(16K) Vt(16K) scores(64K); attn f16 aliases Q+K.
__global__ __launch_bounds__(256) void k_attn(const _Float16* __restrict__ q,
                                              const _Float16* __restrict__ k,
                                              const _Float16* __restrict__ v,
                                              const float* __restrict__ bias,
                                              _Float16* __restrict__ o) {
  extern __shared__ char smem[];
  _Float16* Qs = (_Float16*)smem;            // [s][d]  128x64
  _Float16* Ks = Qs + GS * GDH;              // [s][d]  128x64
  _Float16* Vt = Ks + GS * GDH;              // [d][s]  64x128
  float*    Sc = (float*)(Vt + GDH * GS);    // [q][k]  128x128 fp32
  _Float16* At = Qs;                         // alias over Q+K: [q][k] 128x128 f16

  int tid = threadIdx.x, lane = tid & 31, w = tid >> 5;
  int bb = blockIdx.x, hh = blockIdx.y;

  for (int i = tid; i < GS * GDH; i += 256) {
    int s = i >> 6, d = i & 63;
    size_t g = (((size_t)(bb * GS + s)) * GNH + hh) * GDH + d;
    Qs[i] = q[g];
    Ks[i] = k[g];
    Vt[d * GS + s] = v[g];
  }
  __syncthreads();

  // scores = (Q K^T) * 0.125 + bias  -> Sc
  int r0 = w * 16;
  const float* bp = bias + (size_t)bb * GS * GS;
#pragma unroll
  for (int ct = 0; ct < 8; ++ct) {
    FragF acc; fzero(acc);
#pragma unroll
    for (int ks = 0; ks < 2; ++ks) {
      v16h fa = frag_a(Qs, r0, GDH, ks * 32);
      v16h fb = frag_b(Ks, ct * 16, GDH, ks * 32);
      acc.v = __builtin_amdgcn_wmma_f32_16x16x32_f16(false, fa, false, fb, (short)0, acc.v, false, false);
    }
    int n = ct * 16 + (lane & 15);
#pragma unroll
    for (int i = 0; i < 8; ++i) {
      int m = r0 + ((lane >> 4) << 3) + i;
      Sc[m * GS + n] = acc.f[i] * 0.125f + bp[(size_t)m * GS + n];
    }
  }
  __syncthreads();

  // row softmax (fp32), write f16 probabilities over dead Q/K region
  if (tid < GS) {
    float* row = Sc + (size_t)tid * GS;
    float mx = -1e30f;
    for (int c = 0; c < GS; ++c) mx = fmaxf(mx, row[c]);
    float sm = 0.f;
    for (int c = 0; c < GS; ++c) { float e = __expf(row[c] - mx); row[c] = e; sm += e; }
    float inv = 1.f / sm;
    _Float16* ar = At + (size_t)tid * GS;
    for (int c = 0; c < GS; ++c) ar[c] = (_Float16)(row[c] * inv);
  }
  __syncthreads();

  // O = attn @ V
#pragma unroll
  for (int ct = 0; ct < 4; ++ct) {
    FragF acc; fzero(acc);
#pragma unroll
    for (int ks = 0; ks < 4; ++ks) {
      v16h fa = frag_a(At, r0, GS, ks * 32);
      v16h fb = frag_b(Vt, ct * 16, GS, ks * 32);
      acc.v = __builtin_amdgcn_wmma_f32_16x16x32_f16(false, fa, false, fb, (short)0, acc.v, false, false);
    }
    int d = ct * 16 + (lane & 15);
#pragma unroll
    for (int i = 0; i < 8; ++i) {
      int m = r0 + ((lane >> 4) << 3) + i;
      o[((size_t)(bb * GS + m)) * GH + hh * GDH + d] = (_Float16)acc.f[i];
    }
  }
}

// ---------------- readout: out[b] = h[b,0,:] . W_out + b_out ----------------
__global__ void k_out(const float* __restrict__ h, const float* __restrict__ Wout,
                      const float* __restrict__ bout, float* __restrict__ out) {
  int b = blockIdx.x * blockDim.x + threadIdx.x;
  if (b >= GB) return;
  const float* hr = h + (size_t)b * GS * GH;
  float s = bout[0];
  for (int j = 0; j < GH; ++j) s += hr[j] * Wout[j];
  out[b] = s;
}

// ---------------- host launcher ----------------
extern "C" void kernel_launch(void* const* d_in, const int* in_sizes, int n_in,
                              void* d_out, int out_size, void* d_ws, size_t ws_size,
                              hipStream_t stream) {
  (void)in_sizes; (void)n_in; (void)out_size; (void)ws_size;
  const float* x        = (const float*)d_in[0];
  const float* edgeattr = (const float*)d_in[1];
  const float* W_node   = (const float*)d_in[2];
  const float* b_node   = (const float*)d_in[3];
  const float* W_edge   = (const float*)d_in[4];
  const float* b_edge   = (const float*)d_in[5];
  const float* z_in     = (const float*)d_in[6];
  const float* z_out    = (const float*)d_in[7];
  const float* b_sp     = (const float*)d_in[8];
  const float* e_vec    = (const float*)d_in[9];
  const float* Wq       = (const float*)d_in[10];
  const float* bq       = (const float*)d_in[11];
  const float* Wk       = (const float*)d_in[12];
  const float* bk       = (const float*)d_in[13];
  const float* Wv       = (const float*)d_in[14];
  const float* bv       = (const float*)d_in[15];
  const float* Wo       = (const float*)d_in[16];
  const float* bo       = (const float*)d_in[17];
  const float* ln1g     = (const float*)d_in[18];
  const float* ln1b     = (const float*)d_in[19];
  const float* ln2g     = (const float*)d_in[20];
  const float* ln2b     = (const float*)d_in[21];
  const float* W1       = (const float*)d_in[22];
  const float* b1       = (const float*)d_in[23];
  const float* W2       = (const float*)d_in[24];
  const float* b2       = (const float*)d_in[25];
  const float* W_out    = (const float*)d_in[26];
  const float* b_out    = (const float*)d_in[27];
  const int*   eidx     = (const int*)d_in[28];
  const int*   npath    = (const int*)d_in[30];
  const int*   epath    = (const int*)d_in[31];

  char* ws = (char*)d_ws;
  size_t oH    = 0;                                   // h fp32   32MB
  size_t oHN   = oH   + (size_t)GM * GH * 4;          // hn f16   16MB
  size_t oQ    = oHN  + (size_t)GM * GH * 2;
  size_t oK    = oQ   + (size_t)GM * GH * 2;
  size_t oV    = oK   + (size_t)GM * GH * 2;
  size_t oO    = oV   + (size_t)GM * GH * 2;
  size_t oFFN  = oQ;                                  // alias q..o region (64MB)
  size_t oBias = oO   + (size_t)GM * GH * 2;          // 8MB fp32
  size_t oSS   = oBias + (size_t)GPP * 4;             // step scores
  size_t oX16  = oSS  + (size_t)GE * GLP * 4;         // x f16
  size_t oWN   = oX16 + (size_t)GM * GFN * 2;
  size_t oWQ   = oWN  + (size_t)GFN * GH * 2;
  size_t oWK   = oWQ  + (size_t)GNL * GH * GH * 2;
  size_t oWV   = oWK  + (size_t)GNL * GH * GH * 2;
  size_t oWO   = oWV  + (size_t)GNL * GH * GH * 2;
  size_t oW1   = oWO  + (size_t)GNL * GH * GH * 2;
  size_t oW2   = oW1  + (size_t)GNL * GH * GFFN * 2;
  size_t oDEG  = oW2  + (size_t)GNL * GFFN * GH * 2;  // 2 x 16384 ints
  size_t oWC   = oDEG + (size_t)2 * GM * 4;           // 85 floats

  float*     h    = (float*)(ws + oH);
  _Float16*  hn   = (_Float16*)(ws + oHN);
  _Float16*  qb   = (_Float16*)(ws + oQ);
  _Float16*  kb   = (_Float16*)(ws + oK);
  _Float16*  vb   = (_Float16*)(ws + oV);
  _Float16*  ob   = (_Float16*)(ws + oO);
  _Float16*  ffn  = (_Float16*)(ws + oFFN);
  float*     bias = (float*)(ws + oBias);
  float*     ss   = (float*)(ws + oSS);
  _Float16*  x16  = (_Float16*)(ws + oX16);
  _Float16*  wn16 = (_Float16*)(ws + oWN);
  _Float16*  wq16 = (_Float16*)(ws + oWQ);
  _Float16*  wk16 = (_Float16*)(ws + oWK);
  _Float16*  wv16 = (_Float16*)(ws + oWV);
  _Float16*  wo16 = (_Float16*)(ws + oWO);
  _Float16*  w116 = (_Float16*)(ws + oW1);
  _Float16*  w216 = (_Float16*)(ws + oW2);
  int*       indeg  = (int*)(ws + oDEG);
  int*       outdeg = indeg + GM;
  float*     wc   = (float*)(ws + oWC);

  auto cvt = [&](const float* s, _Float16* d, int n) {
    k_f32_to_f16<<<(n + 255) / 256, 256, 0, stream>>>(s, d, n);
  };
  cvt(W_node, wn16, GFN * GH);
  cvt(Wq, wq16, GNL * GH * GH);
  cvt(Wk, wk16, GNL * GH * GH);
  cvt(Wv, wv16, GNL * GH * GH);
  cvt(Wo, wo16, GNL * GH * GH);
  cvt(W1, w116, GNL * GH * GFFN);
  cvt(W2, w216, GNL * GFFN * GH);

  // degrees
  k_zero_i32<<<(2 * GM + 255) / 256, 256, 0, stream>>>(indeg, 2 * GM);
  k_degrees<<<(GE + 255) / 256, 256, 0, stream>>>(eidx, outdeg, indeg);

  // node embedding
  k_build_x16<<<(GM * GFN + 255) / 256, 256, 0, stream>>>(x, x16);
  k_gemm<false, 0, false><<<dim3(GM / 128, GH / 128), 256, 0, stream>>>(
      x16, wn16, b_node, nullptr, h, nullptr, GM, GH, GFN);
  k_degadd<<<(GM * GH + 255) / 256, 256, 0, stream>>>(h, indeg, outdeg, z_in, z_out);

  // structural bias
  k_wc<<<1, 128, 0, stream>>>(W_edge, b_edge, e_vec, wc);
  k_stepscores<<<(GE + 255) / 256, 256, 0, stream>>>(edgeattr, wc, ss);
  k_bias<<<(GPP + 255) / 256, 256, 0, stream>>>(npath, epath, ss, b_sp, bias);

  const size_t attn_lds = (size_t)(3 * GS * GDH) * 2 + (size_t)GS * GS * 4; // 114688 B

  for (int i = 0; i < GNL; ++i) {
    const _Float16* wqi = wq16 + (size_t)i * GH * GH;
    const _Float16* wki = wk16 + (size_t)i * GH * GH;
    const _Float16* wvi = wv16 + (size_t)i * GH * GH;
    const _Float16* woi = wo16 + (size_t)i * GH * GH;
    const _Float16* w1i = w116 + (size_t)i * GH * GFFN;
    const _Float16* w2i = w216 + (size_t)i * GFFN * GH;

    k_ln<<<GM, 256, 0, stream>>>(h, ln1g + i * GH, ln1b + i * GH, hn);
    k_gemm<false, 0, true><<<dim3(GM / 128, GH / 128), 256, 0, stream>>>(
        hn, wqi, bq + i * GH, nullptr, nullptr, qb, GM, GH, GH);
    k_gemm<false, 0, true><<<dim3(GM / 128, GH / 128), 256, 0, stream>>>(
        hn, wki, bk + i * GH, nullptr, nullptr, kb, GM, GH, GH);
    k_gemm<false, 0, true><<<dim3(GM / 128, GH / 128), 256, 0, stream>>>(
        hn, wvi, bv + i * GH, nullptr, nullptr, vb, GM, GH, GH);
    k_attn<<<dim3(GB, GNH), 256, attn_lds, stream>>>(qb, kb, vb, bias, ob);
    k_gemm<true, 0, false><<<dim3(GM / 128, GH / 128), 256, 0, stream>>>(
        ob, woi, bo + i * GH, h, h, nullptr, GM, GH, GH);
    k_ln<<<GM, 256, 0, stream>>>(h, ln2g + i * GH, ln2b + i * GH, hn);
    k_gemm<false, 1, true><<<dim3(GM / 128, GFFN / 128), 256, 0, stream>>>(
        hn, w1i, b1 + i * GFFN, nullptr, nullptr, ffn, GM, GFFN, GH);
    k_gemm<true, 0, false><<<dim3(GM / 128, GH / 128), 256, 0, stream>>>(
        ffn, w2i, b2 + i * GH, h, h, nullptr, GM, GH, GFFN);
  }

  k_out<<<1, 128, 0, stream>>>(h, W_out, b_out, (float*)d_out);
}
